// AttentionBlock_22273700397509
// MI455X (gfx1250) — matrix operand order repaired and checked by reading, not compiled, not run on hardware
//
#include <hip/hip_runtime.h>

typedef __attribute__((ext_vector_type(16))) _Float16 v16h;
typedef __attribute__((ext_vector_type(8)))  _Float16 v8h;
typedef __attribute__((ext_vector_type(8)))  float    v8f;

#define N_CH   512
#define DK     64
#define NH     8
#define S_LEN  1024
#define BATCH  16
#define ATT_SCALE 0.35355339059327373f   // 64^-0.25
#define PN_EPS 1e-8f
#define NEG_BIG (-3.402823466e+38f)

// ---------------- CDNA5 async global->LDS copy (ASYNCcnt-tracked) ----------------
// Each lane moves 16B from its global address to its LDS address (ISA §15.18.3 op 98).
static __device__ __forceinline__ void async_g2l_b128(_Float16* lds_dst,
                                                      const _Float16* gsrc) {
#if defined(__gfx1250__)
  unsigned int l = (unsigned int)(uintptr_t)lds_dst;   // generic->LDS: addr[31:0]
  asm volatile("global_load_async_to_lds_b128 %0, %1, off"
               :: "v"(l), "v"(gsrc) : "memory");
#else
  *(v8h*)lds_dst = *(const v8h*)gsrc;
#endif
}

static __device__ __forceinline__ void wait_async0() {
#if defined(__gfx1250__)
  asm volatile("s_wait_asynccnt 0" ::: "memory");
#endif
}

// ---------------- WMMA fragment helpers (contiguous k order: lane half h holds k = 16h..16h+15 in A and B) ----------------

static __device__ __forceinline__ v8f wmma_f16(v16h a, v16h b, v8f c) {
  return __builtin_amdgcn_wmma_f32_16x16x32_f16(false, a, false, b, (short)0, c,
                                                false, false);
}

// A-matrix 16x32 f16 from row-major [16 x ld]:
// lanes 0-15 hold row M=lane, K=0..15;
// lanes 16-31 hold row M=lane-16, K=16..31.
static __device__ __forceinline__ v16h frag_a(const _Float16* p, int ld) {
  int lane = threadIdx.x & 31;
  return *(const v16h*)(p + (lane & 15) * ld + ((lane >> 4) * 16));
}

// B-matrix 32x16 f16 where B[k][n] = W[n][k], W row-major [16 x ld]:
// lanes 0-15: column N=lane, K=0..15; lanes 16-31: column N=lane-16, K=16..31.
static __device__ __forceinline__ v16h frag_b(const _Float16* p, int ld) {
  int lane = threadIdx.x & 31;
  return *(const v16h*)(p + (lane & 15) * ld + ((lane >> 4) * 16));
}

// ---------------- prep kernels ----------------

__global__ void k_f32_to_f16(const float* __restrict__ src,
                             _Float16* __restrict__ dst, int n) {
  int i = blockIdx.x * blockDim.x + threadIdx.x;
  if (i < n) dst[i] = (_Float16)src[i];
}

// x [B, C, S] f32 -> xt [B*S, C] f16  (LDS-tiled transpose, 32x32 tiles)
__global__ __launch_bounds__(256) void k_transpose_x(const float* __restrict__ x,
                                                     _Float16* __restrict__ xt) {
  __shared__ _Float16 tile[32][33];
  int b  = blockIdx.z;
  int c0 = blockIdx.y * 32;
  int s0 = blockIdx.x * 32;
  int tx = threadIdx.x;          // 0..31 along s (coalesced read)
  int ty = threadIdx.y;          // 0..7
#pragma unroll
  for (int i = 0; i < 4; ++i) {
    int c = c0 + ty + i * 8;
    tile[ty + i * 8][tx] = (_Float16)x[((size_t)b * N_CH + c) * S_LEN + s0 + tx];
  }
  __syncthreads();
#pragma unroll
  for (int i = 0; i < 4; ++i) {
    int s = s0 + ty + i * 8;
    xt[((size_t)b * S_LEN + s) * N_CH + c0 + tx] = tile[tx][ty + i * 8];
  }
}

// ---------------- kernel 1: QKV projection + pixnorm + scale ----------------
// grid (M/16 = 1024, heads = 8), 384 threads = 12 waves.
// A strip (16 tokens x 512 ch, 16KB) staged once in LDS via async copies and
// shared by all 12 waves. Wave w owns output channels [head*192 + w*16, +16).
__global__ __launch_bounds__(384) void k_qkv(const _Float16* __restrict__ xt,
                                             const _Float16* __restrict__ wproj,
                                             _Float16* __restrict__ Qh,
                                             _Float16* __restrict__ Kh,
                                             _Float16* __restrict__ Vt) {
  __shared__ __align__(64) _Float16 aTile[16 * N_CH];   // 16 KB
  __shared__ float tile[16][192];
  __shared__ float rnorm[16][3];
  int wave = threadIdx.x >> 5;
  int lane = threadIdx.x & 31;
  int m0   = blockIdx.x * 16;          // token tile
  int head = blockIdx.y;
  int n0   = head * 192 + wave * 16;   // output-channel tile
  int t    = threadIdx.x;

  // stage A strip: 1024 x 16B chunks over 384 threads
  {
    const _Float16* g = xt + (size_t)m0 * N_CH;
    for (int c = t; c < 1024; c += 384)
      async_g2l_b128(aTile + c * 8, g + c * 8);
    wait_async0();
  }
  __syncthreads();

  v8f acc = {};
  const _Float16* bp = wproj + (size_t)n0 * N_CH;
#pragma unroll
  for (int kk = 0; kk < N_CH; kk += 32) {
    v16h a = frag_a(aTile + kk, N_CH);
    v16h b = frag_b(bp + kk, N_CH);
    acc = wmma_f16(a, b, acc);
  }

  int rb = (lane >> 4) * 8;   // row base (C/D layout)
  int cl = lane & 15;         // column within tile
#pragma unroll
  for (int r = 0; r < 8; ++r)
    tile[r + rb][wave * 16 + cl] = acc[r];
  __syncthreads();

  if (t < 48) {               // 16 rows x 3 groups (q,k,v): RMS per 64 channels
    int row = t / 3, g = t % 3;
    float s = 0.f;
#pragma unroll 8
    for (int c = 0; c < 64; ++c) {
      float v = tile[row][g * 64 + c];
      s += v * v;
    }
    float rn = rsqrtf(s * (1.0f / 64.0f) + PN_EPS);
    if (g < 2) rn *= ATT_SCALE;     // bake softmax scale into q and k
    rnorm[row][g] = rn;
  }
  __syncthreads();

  int bb = m0 / S_LEN, s0 = m0 % S_LEN;
  int bh = bb * NH + head;
#pragma unroll
  for (int e = 0; e < 8; ++e) {     // 3072 elements / 384 threads
    int lin = e * 384 + t;
    int row = lin / 192, col = lin % 192;
    int g = col / 64, d = col % 64;
    _Float16 hv = (_Float16)(tile[row][col] * rnorm[row][g]);
    int s = s0 + row;
    if (g == 0)      Qh[((size_t)bh * S_LEN + s) * DK + d] = hv;
    else if (g == 1) Kh[((size_t)bh * S_LEN + s) * DK + d] = hv;
    else             Vt[((size_t)bh * DK + d) * S_LEN + s] = hv;   // transposed
  }
}

// ---------------- kernel 2: flash attention per (b,h) ----------------
// grid (B*H = 128, S/128 = 8), 256 threads = 8 waves; wave owns 16 query rows.
// Per 32-key step, K tile (32x64) and V tile (64x32, from Vt) are staged in LDS
// once per block with async copies; all 8 waves consume them.
__global__ __launch_bounds__(256) void k_attn(const _Float16* __restrict__ Qh,
                                              const _Float16* __restrict__ Kh,
                                              const _Float16* __restrict__ Vt,
                                              _Float16* __restrict__ resh) {
  __shared__ __align__(64) _Float16 kTile[32 * DK];    // K[j0..j0+31][0..63], 4KB
  __shared__ __align__(64) _Float16 vTile[DK * 32];    // Vt[d][j0..j0+31],    4KB
  __shared__ __align__(64) _Float16 lds_p[8][16 * 32]; // per-wave P tiles,    8KB
  int wave = threadIdx.x >> 5;
  int lane = threadIdx.x & 31;
  int tid  = threadIdx.x;
  int bh = blockIdx.x;
  int b = bh >> 3, h = bh & 7;
  int m0 = blockIdx.y * 128 + wave * 16;    // query row within (b,h)

  const _Float16* Q = Qh + (size_t)bh * S_LEN * DK;
  const _Float16* K = Kh + (size_t)bh * S_LEN * DK;
  const _Float16* V = Vt + (size_t)bh * DK * S_LEN;   // [64][1024]

  v8f o0 = {}, o1 = {}, o2 = {}, o3 = {};
  float mrow[8], lrow[8];
#pragma unroll
  for (int r = 0; r < 8; ++r) { mrow[r] = NEG_BIG; lrow[r] = 0.f; }

  // Q fragments are loop-invariant: contraction d 0..31 and 32..63
  v16h qa0 = frag_a(Q + (size_t)m0 * DK + 0, DK);
  v16h qa1 = frag_a(Q + (size_t)m0 * DK + 32, DK);

  _Float16* P = lds_p[wave];
  int rb = (lane >> 4) * 8;
  int cl = lane & 15;
  int vd = tid >> 2, vp = tid & 3;          // V-tile chunk: row d, quarter

  for (int j0 = 0; j0 < S_LEN; j0 += 32) {
    __syncthreads();   // previous iteration's LDS reads are done
    // K tile: 32 rows x 64 halves, contiguous 4KB -> 256 b128 chunks
    async_g2l_b128(kTile + tid * 8, K + (size_t)j0 * DK + tid * 8);
    // V tile: 64 rows of 32 halves from Vt[d][j0..]
    async_g2l_b128(vTile + vd * 32 + vp * 8, V + (size_t)vd * S_LEN + j0 + vp * 8);
    wait_async0();
    __syncthreads();

    // S = Q (16x64) x K^T (64x32) -> two 16x16 tiles
    v8f s0 = {}, s1 = {};
    {
      v16h b0 = frag_b(kTile + 0 * DK + 0, DK);
      v16h b1 = frag_b(kTile + 16 * DK + 0, DK);
      s0 = wmma_f16(qa0, b0, s0);
      s1 = wmma_f16(qa0, b1, s1);
      b0 = frag_b(kTile + 0 * DK + 32, DK);
      b1 = frag_b(kTile + 16 * DK + 32, DK);
      s0 = wmma_f16(qa1, b0, s0);
      s1 = wmma_f16(qa1, b1, s1);
    }
    // online softmax over the 32 new columns (half-wave row reductions)
#pragma unroll
    for (int r = 0; r < 8; ++r) {
      float v0 = s0[r], v1 = s1[r];
      float mx = fmaxf(v0, v1);
#pragma unroll
      for (int off = 8; off > 0; off >>= 1)
        mx = fmaxf(mx, __shfl_xor(mx, off, 32));
      float m_new = fmaxf(mrow[r], mx);
      float sc = __expf(mrow[r] - m_new);
      float p0 = __expf(v0 - m_new);
      float p1 = __expf(v1 - m_new);
      float ps = p0 + p1;
#pragma unroll
      for (int off = 8; off > 0; off >>= 1)
        ps += __shfl_xor(ps, off, 32);
      lrow[r] = lrow[r] * sc + ps;
      mrow[r] = m_new;
      o0[r] *= sc; o1[r] *= sc; o2[r] *= sc; o3[r] *= sc;
      int row = r + rb;
      P[row * 32 + cl]      = (_Float16)p0;
      P[row * 32 + 16 + cl] = (_Float16)p1;
    }
    // O += P (16x32) x V (32x64): reshape P D-layout -> A-layout via LDS
    {
      v16h pa = *(const v16h*)(P + (lane & 15) * 32 + ((lane >> 4) * 16));
      v16h vb;
      vb = frag_b(vTile + 0  * 32, 32); o0 = wmma_f16(pa, vb, o0);
      vb = frag_b(vTile + 16 * 32, 32); o1 = wmma_f16(pa, vb, o1);
      vb = frag_b(vTile + 32 * 32, 32); o2 = wmma_f16(pa, vb, o2);
      vb = frag_b(vTile + 48 * 32, 32); o3 = wmma_f16(pa, vb, o3);
    }
  }

  // normalize and store res [B, S, C] with c = h*64 + d
#pragma unroll
  for (int r = 0; r < 8; ++r) {
    float inv = 1.0f / lrow[r];
    int row = r + rb;
    int s = m0 + row;
    _Float16* dst = resh + ((size_t)(b * S_LEN + s)) * N_CH + h * DK;
    dst[0  + cl] = (_Float16)(o0[r] * inv);
    dst[16 + cl] = (_Float16)(o1[r] * inv);
    dst[32 + cl] = (_Float16)(o2[r] * inv);
    dst[48 + cl] = (_Float16)(o3[r] * inv);
  }
}

// ---------------- kernel 3: out projection + residual ----------------
// grid (M/16 = 1024, 4), 256 threads = 8 waves; A strip staged in LDS.
__global__ __launch_bounds__(256) void k_out(const _Float16* __restrict__ resh,
                                             const _Float16* __restrict__ wout,
                                             const float* __restrict__ x,
                                             float* __restrict__ out) {
  __shared__ __align__(64) _Float16 aTile[16 * N_CH];  // 16 KB
  int wave = threadIdx.x >> 5;
  int lane = threadIdx.x & 31;
  int t    = threadIdx.x;
  int m0 = blockIdx.x * 16;
  int n0 = (blockIdx.y * 8 + wave) * 16;

  {
    const _Float16* g = resh + (size_t)m0 * N_CH;
    for (int c = t; c < 1024; c += 256)
      async_g2l_b128(aTile + c * 8, g + c * 8);
    wait_async0();
  }
  __syncthreads();

  v8f acc = {};
  const _Float16* bp = wout + (size_t)n0 * N_CH;
#pragma unroll
  for (int kk = 0; kk < N_CH; kk += 32) {
    v16h a = frag_a(aTile + kk, N_CH);
    v16h b = frag_b(bp + kk, N_CH);
    acc = wmma_f16(a, b, acc);
  }

  int bb = m0 / S_LEN, s0 = m0 % S_LEN;
  int rb = (lane >> 4) * 8, cl = lane & 15;
  int c = n0 + cl;
#pragma unroll
  for (int r = 0; r < 8; ++r) {
    size_t idx = (size_t)bb * (N_CH * S_LEN) + (size_t)c * S_LEN + s0 + r + rb;
    out[idx] = x[idx] + acc[r];
  }
}

// ---------------- launch ----------------

extern "C" void kernel_launch(void* const* d_in, const int* in_sizes, int n_in,
                              void* d_out, int out_size, void* d_ws, size_t ws_size,
                              hipStream_t stream) {
  const float* x      = (const float*)d_in[0];   // [16, 512, 32, 32]
  const float* w_proj = (const float*)d_in[1];   // [1536, 512]
  const float* w_out  = (const float*)d_in[2];   // [512, 512]
  float* out = (float*)d_out;

  char* ws = (char*)d_ws;
  size_t off = 0;
  auto carve = [&](size_t bytes) {
    void* p = ws + off;
    off += (bytes + 255) & ~(size_t)255;
    return p;
  };
  const size_t TOK = (size_t)BATCH * S_LEN;               // 16384
  _Float16* xt   = (_Float16*)carve(TOK * N_CH * 2);      // 16 MB
  _Float16* wpj  = (_Float16*)carve((size_t)3 * N_CH * N_CH * 2);
  _Float16* wo   = (_Float16*)carve((size_t)N_CH * N_CH * 2);
  _Float16* Qh   = (_Float16*)carve(TOK * N_CH * 2);      // [B,H,S,64]
  _Float16* Kh   = (_Float16*)carve(TOK * N_CH * 2);
  _Float16* Vt   = (_Float16*)carve(TOK * N_CH * 2);      // [B,H,64,S]
  _Float16* resh = (_Float16*)carve(TOK * N_CH * 2);      // [B,S,C]

  int nwp = 3 * N_CH * N_CH;
  int nwo = N_CH * N_CH;
  k_f32_to_f16<<<(nwp + 255) / 256, 256, 0, stream>>>(w_proj, wpj, nwp);
  k_f32_to_f16<<<(nwo + 255) / 256, 256, 0, stream>>>(w_out, wo, nwo);
  k_transpose_x<<<dim3(S_LEN / 32, N_CH / 32, BATCH), dim3(32, 8), 0, stream>>>(x, xt);
  k_qkv<<<dim3(TOK / 16, NH), 384, 0, stream>>>(xt, wpj, Qh, Kh, Vt);
  k_attn<<<dim3(BATCH * NH, S_LEN / 128), 256, 0, stream>>>(Qh, Kh, Vt, resh);
  k_out<<<dim3(TOK / 16, N_CH / (8 * 16)), 256, 0, stream>>>(resh, wo, x, out);
}
